// ExternalAttention2_82952998355251
// MI455X (gfx1250) — matrix operand order repaired and checked by reading, not compiled, not run on hardware
//
#include <hip/hip_runtime.h>
#include <hip/hip_bf16.h>

typedef __bf16 bf16_t;
typedef __attribute__((ext_vector_type(16))) __bf16 v16bf;
typedef __attribute__((ext_vector_type(8)))  __bf16 v8bf;
typedef __attribute__((ext_vector_type(8)))  float  v8f;

union v16u { v16bf v; v8bf h[2]; };

#define E_DIM   1024
#define M_SLOTS 256
#define N_ROWS  32768

// ---- CDNA5 async global->LDS copy (ASYNCcnt-tracked, bypasses VGPRs) -------
template<typename T>
__device__ __forceinline__ uint32_t lds_addr(T* p) {
  return (uint32_t)(uintptr_t)(__attribute__((address_space(3))) T*)p;
}
__device__ __forceinline__ void async_load_b128(uint32_t lds_off, const void* gptr) {
  asm volatile("global_load_async_to_lds_b128 %0, %1, off"
               :: "v"(lds_off), "v"(gptr) : "memory");
}
__device__ __forceinline__ void wait_async0() {
  asm volatile("s_wait_asynccnt 0x0" ::: "memory");
}

// A-fragment (16x32 bf16): lane half h holds K h*16 .. h*16+15 of its row
__device__ __forceinline__ v16bf frag_a16(const bf16_t* rowp, int lhalf) {
  v16u u;
  u.h[0] = *(const v8bf*)(rowp + lhalf * 16);
  u.h[1] = *(const v8bf*)(rowp + lhalf * 16 + 8);
  return u.v;
}
// B-fragment (32x16 bf16): lane half h holds K h*16 .. h*16+15 of its column
__device__ __forceinline__ v16bf frag_b16(const bf16_t* colp, int lhalf) {
  v16u u;
  u.h[0] = *(const v8bf*)(colp + lhalf * 16);
  u.h[1] = *(const v8bf*)(colp + lhalf * 16 + 8);
  return u.v;
}

__device__ __forceinline__ v8f wmma_bf16(v16bf a, v16bf b, v8f c) {
  return __builtin_amdgcn_wmma_f32_16x16x32_bf16(false, a, false, b, (short)0, c,
                                                 false, false);
}

// ---------------------------------------------------------------------------
// out[m,n] = sum_k A[m,k] * W[n,k] + bias[n] (+ resid[m,n] if MODE==1)
// MODE 0: out bf16; MODE 1: out f32 with residual.
// AF32: A is f32 (register-staged + converted); else bf16 (async-staged).
// Block: 256 threads, tile 128(M) x 128(N), K-step 32, double-buffered LDS.
// ---------------------------------------------------------------------------
template<int MODE, bool AF32>
__launch_bounds__(256, 2)
__global__ void gemm_bf16(const void* __restrict__ Ap,
                          const bf16_t* __restrict__ W,
                          const float* __restrict__ bias,
                          const float* __restrict__ resid,
                          void* __restrict__ outp,
                          int M, int N, int K)
{
  const int tid   = threadIdx.x;
  const int lane  = tid & 31;
  const int wave  = tid >> 5;      // 0..7
  const int wm    = wave >> 2;     // 0..1 (64 rows each)
  const int wn    = wave & 3;      // 0..3 (32 cols each)
  const int lrow  = lane & 15;
  const int lhalf = lane >> 4;
  const int mBase = blockIdx.y * 128;
  const int nBase = blockIdx.x * 128;

  __shared__ bf16_t sA[2][128][48];   // stride 48 elems (96B, 16B aligned)
  __shared__ bf16_t sB[2][128][48];

  // 16B-chunk mapping for async staging of a 128x32 tile (512 chunks)
  const int r0 = tid >> 2,          p0 = (tid & 3) * 8;          // chunk tid
  const int r1 = (tid + 256) >> 2,  p1 = ((tid + 256) & 3) * 8;  // chunk tid+256

  // f32-A register staging mapping
  const int ldr = tid >> 1;        // 0..127
  const int ldc = (tid & 1) * 16;  // 0 / 16

  const v8f vzero = {0.f,0.f,0.f,0.f,0.f,0.f,0.f,0.f};
  v8f acc[4][2];
  for (int i = 0; i < 4; ++i)
    for (int j = 0; j < 2; ++j) acc[i][j] = vzero;

  const float*  Af = (const float*)Ap;
  const bf16_t* Ab = (const bf16_t*)Ap;

  float4 fa0, fa1, fa2, fa3;

  // ---- prologue: stage K-step 0 into buffer 0 ----
  async_load_b128(lds_addr(&sB[0][r0][p0]), W + (size_t)(nBase + r0) * K + p0);
  async_load_b128(lds_addr(&sB[0][r1][p1]), W + (size_t)(nBase + r1) * K + p1);
  if (AF32) {
    const float* p = Af + (size_t)(mBase + ldr) * K + ldc;
    fa0 = *(const float4*)(p + 0);
    fa1 = *(const float4*)(p + 4);
    fa2 = *(const float4*)(p + 8);
    fa3 = *(const float4*)(p + 12);
    float ftmp[16];
    *(float4*)&ftmp[0]  = fa0; *(float4*)&ftmp[4]  = fa1;
    *(float4*)&ftmp[8]  = fa2; *(float4*)&ftmp[12] = fa3;
    v8bf h0, h1;
    for (int e = 0; e < 8; ++e) { h0[e] = (bf16_t)ftmp[e]; h1[e] = (bf16_t)ftmp[e + 8]; }
    *(v8bf*)&sA[0][ldr][ldc]     = h0;
    *(v8bf*)&sA[0][ldr][ldc + 8] = h1;
  } else {
    async_load_b128(lds_addr(&sA[0][r0][p0]), Ab + (size_t)(mBase + r0) * K + p0);
    async_load_b128(lds_addr(&sA[0][r1][p1]), Ab + (size_t)(mBase + r1) * K + p1);
  }
  wait_async0();
  __syncthreads();

  const int nk = K >> 5;
  for (int ks = 0; ks < nk; ++ks) {
    const int cur = ks & 1, nxt = cur ^ 1;
    const bool hasNext = (ks + 1 < nk);
    if (hasNext) {   // issue async copies for tile ks+1; they overlap the WMMAs below
      const int k0 = (ks + 1) << 5;
      async_load_b128(lds_addr(&sB[nxt][r0][p0]), W + (size_t)(nBase + r0) * K + k0 + p0);
      async_load_b128(lds_addr(&sB[nxt][r1][p1]), W + (size_t)(nBase + r1) * K + k0 + p1);
      if (AF32) {
        const float* p = Af + (size_t)(mBase + ldr) * K + k0 + ldc;
        fa0 = *(const float4*)(p + 0);
        fa1 = *(const float4*)(p + 4);
        fa2 = *(const float4*)(p + 8);
        fa3 = *(const float4*)(p + 12);
      } else {
        async_load_b128(lds_addr(&sA[nxt][r0][p0]), Ab + (size_t)(mBase + r0) * K + k0 + p0);
        async_load_b128(lds_addr(&sA[nxt][r1][p1]), Ab + (size_t)(mBase + r1) * K + k0 + p1);
      }
    }

    v16bf aF[4], bF[2];
    for (int i = 0; i < 4; ++i)
      aF[i] = frag_a16(&sA[cur][wm * 64 + i * 16 + lrow][0], lhalf);
    for (int j = 0; j < 2; ++j)
      bF[j] = frag_b16(&sB[cur][wn * 32 + j * 16 + lrow][0], lhalf);
    for (int i = 0; i < 4; ++i)
      for (int j = 0; j < 2; ++j)
        acc[i][j] = wmma_bf16(aF[i], bF[j], acc[i][j]);

    if (hasNext) {
      if (AF32) {      // convert staged f32 -> bf16 into next buffer
        float ftmp[16];
        *(float4*)&ftmp[0]  = fa0; *(float4*)&ftmp[4]  = fa1;
        *(float4*)&ftmp[8]  = fa2; *(float4*)&ftmp[12] = fa3;
        v8bf h0, h1;
        for (int e = 0; e < 8; ++e) { h0[e] = (bf16_t)ftmp[e]; h1[e] = (bf16_t)ftmp[e + 8]; }
        *(v8bf*)&sA[nxt][ldr][ldc]     = h0;
        *(v8bf*)&sA[nxt][ldr][ldc + 8] = h1;
      }
      wait_async0();
      __syncthreads();
    }
  }

  // epilogue: D layout — lane holds column lrow, rows lhalf*8 + r
  for (int j = 0; j < 2; ++j) {
    const int col = nBase + wn * 32 + j * 16 + lrow;
    const float b = bias[col];
    for (int i = 0; i < 4; ++i) {
      const int row0 = mBase + wm * 64 + i * 16 + lhalf * 8;
      for (int r = 0; r < 8; ++r) {
        const size_t idx = (size_t)(row0 + r) * N + col;
        const float v = acc[i][j][r] + b;
        if (MODE == 0) ((bf16_t*)outp)[idx] = (bf16_t)v;
        else           ((float*)outp)[idx]  = v + resid[idx];
      }
    }
  }
}

// ---------------------------------------------------------------------------
// Fused: scores = Q*mem^T -> softmax -> attended = weights*mem (bf16 out)
// One block = 64 query rows; mem/memT (0.5MB each) are L2-resident.
// ---------------------------------------------------------------------------
__launch_bounds__(256, 2)
__global__ void attn_softmax(const bf16_t* __restrict__ Q,
                             const bf16_t* __restrict__ Mem,
                             const bf16_t* __restrict__ MemT,
                             bf16_t* __restrict__ Att)
{
  const int tid   = threadIdx.x;
  const int lane  = tid & 31;
  const int wave  = tid >> 5;
  const int lrow  = lane & 15;
  const int lhalf = lane >> 4;
  const int rowBase = blockIdx.x * 64;

  __shared__ float  sS[64][260];     // scores f32, row stride 1040B (16B mult)
  __shared__ bf16_t sQ[2][64][48];   // double-buffered Q K-step tiles
  bf16_t* sWb = (bf16_t*)&sS[0][0];  // softmax-weight overlay, stride 520 bf16

  const v8f vzero = {0.f,0.f,0.f,0.f,0.f,0.f,0.f,0.f};
  v8f acc[4][2];
  for (int i = 0; i < 4; ++i)
    for (int j = 0; j < 2; ++j) acc[i][j] = vzero;

  // ---- Phase A: scores[64,256] = Q[64,1024] x mem^T ----
  const int qr = tid >> 2;          // 0..63
  const int qc = (tid & 3) * 8;     // 0,8,16,24

  async_load_b128(lds_addr(&sQ[0][qr][qc]), Q + (size_t)(rowBase + qr) * E_DIM + qc);
  wait_async0();
  __syncthreads();

  for (int ks = 0; ks < 32; ++ks) {
    const int cur = ks & 1;
    const bool hasNext = (ks + 1 < 32);
    if (hasNext)
      async_load_b128(lds_addr(&sQ[cur ^ 1][qr][qc]),
                      Q + (size_t)(rowBase + qr) * E_DIM + (ks + 1) * 32 + qc);

    const int k0 = ks * 32;
    v16bf aF[4];
    for (int i = 0; i < 4; ++i)
      aF[i] = frag_a16(&sQ[cur][i * 16 + lrow][0], lhalf);
    for (int j = 0; j < 2; ++j) {
      const int slot = wave * 32 + j * 16 + lrow;
      v16bf bF = frag_b16(Mem + (size_t)slot * E_DIM + k0, lhalf);
      for (int i = 0; i < 4; ++i)
        acc[i][j] = wmma_bf16(aF[i], bF, acc[i][j]);
    }
    if (hasNext) {
      wait_async0();
      __syncthreads();
    }
  }

  for (int i = 0; i < 4; ++i)
    for (int j = 0; j < 2; ++j)
      for (int r = 0; r < 8; ++r)
        sS[i * 16 + lhalf * 8 + r][wave * 32 + j * 16 + lrow] = acc[i][j][r];
  __syncthreads();

  // ---- softmax: one thread per row; weights written bf16 in-place ----
  if (tid < 64) {
    float mx = -3.0e38f;
    for (int c = 0; c < 256; ++c) mx = fmaxf(mx, sS[tid][c]);
    float sum = 0.f;
    for (int c = 0; c < 256; ++c) sum += __expf(sS[tid][c] - mx);
    const float inv = 1.f / sum;
    bf16_t* wr = sWb + (size_t)tid * 520;
    for (int c = 0; c < 256; ++c)   // write@2c always behind read@4c: safe
      wr[c] = (bf16_t)(__expf(sS[tid][c] - mx) * inv);
  }
  __syncthreads();

  // ---- Phase B: attended[64,1024] = weights[64,256] x mem, 4 column passes --
  for (int p = 0; p < 4; ++p) {
    const int colBase = p * 256 + wave * 32;
    v8f acc2[4][2];
    for (int i = 0; i < 4; ++i)
      for (int j = 0; j < 2; ++j) acc2[i][j] = vzero;

    for (int ks = 0; ks < 8; ++ks) {  // K = 256
      v16bf aF[4];
      for (int i = 0; i < 4; ++i)
        aF[i] = frag_a16(sWb + (size_t)(i * 16 + lrow) * 520 + ks * 32, lhalf);
      for (int j = 0; j < 2; ++j) {
        const int n = colBase + j * 16 + lrow;
        v16bf bF = frag_b16(MemT + (size_t)n * M_SLOTS + ks * 32, lhalf);
        for (int i = 0; i < 4; ++i)
          acc2[i][j] = wmma_bf16(aF[i], bF, acc2[i][j]);
      }
    }
    for (int j = 0; j < 2; ++j) {
      const int col = colBase + j * 16 + lrow;
      for (int i = 0; i < 4; ++i) {
        const int row0 = rowBase + i * 16 + lhalf * 8;
        for (int r = 0; r < 8; ++r)
          Att[(size_t)(row0 + r) * E_DIM + col] = (bf16_t)acc2[i][j][r];
      }
    }
  }
}

// ---------------------------------------------------------------------------
__global__ void cvt_kernel(const float* __restrict__ in, bf16_t* __restrict__ out,
                           int n) {
  int i = blockIdx.x * blockDim.x + threadIdx.x;
  if (i < n) out[i] = (bf16_t)in[i];
}

// mem [256,1024] f32 -> memT [1024,256] bf16
__global__ void transpose_cvt_kernel(const float* __restrict__ in,
                                     bf16_t* __restrict__ out) {
  int i = blockIdx.x * blockDim.x + threadIdx.x;   // r*1024 + c
  int r = i >> 10, c = i & 1023;
  out[(size_t)c * M_SLOTS + r] = (bf16_t)in[i];
}

// ---------------------------------------------------------------------------
extern "C" void kernel_launch(void* const* d_in, const int* in_sizes, int n_in,
                              void* d_out, int out_size, void* d_ws, size_t ws_size,
                              hipStream_t stream)
{
  const float* x   = (const float*)d_in[0];
  const float* mem = (const float*)d_in[1];
  const float* Wq  = (const float*)d_in[2];
  const float* bq  = (const float*)d_in[3];
  const float* Wo  = (const float*)d_in[4];
  const float* bo  = (const float*)d_in[5];
  float* out = (float*)d_out;

  char* ws = (char*)d_ws;
  size_t off = 0;
  auto walloc = [&](size_t bytes) {
    char* p = ws + off;
    off += (bytes + 255) & ~(size_t)255;
    return p;
  };
  bf16_t* qbf   = (bf16_t*)walloc((size_t)N_ROWS * E_DIM * 2);   // 64 MB
  bf16_t* att   = (bf16_t*)walloc((size_t)N_ROWS * E_DIM * 2);   // 64 MB
  bf16_t* wqbf  = (bf16_t*)walloc((size_t)E_DIM * E_DIM * 2);    // 2 MB
  bf16_t* wobf  = (bf16_t*)walloc((size_t)E_DIM * E_DIM * 2);    // 2 MB
  bf16_t* membf = (bf16_t*)walloc((size_t)M_SLOTS * E_DIM * 2);  // 0.5 MB
  bf16_t* memT  = (bf16_t*)walloc((size_t)M_SLOTS * E_DIM * 2);  // 0.5 MB

  cvt_kernel<<<(E_DIM * E_DIM) / 256, 256, 0, stream>>>(Wq, wqbf, E_DIM * E_DIM);
  cvt_kernel<<<(E_DIM * E_DIM) / 256, 256, 0, stream>>>(Wo, wobf, E_DIM * E_DIM);
  cvt_kernel<<<(M_SLOTS * E_DIM) / 256, 256, 0, stream>>>(mem, membf,
                                                          M_SLOTS * E_DIM);
  transpose_cvt_kernel<<<(M_SLOTS * E_DIM) / 256, 256, 0, stream>>>(mem, memT);

  // queries = x * Wq^T + bq   (bf16 out)
  gemm_bf16<0, true><<<dim3(E_DIM / 128, N_ROWS / 128), 256, 0, stream>>>(
      x, wqbf, bq, nullptr, qbf, N_ROWS, E_DIM, E_DIM);

  // attention + softmax + attended (bf16 out)
  attn_softmax<<<N_ROWS / 64, 256, 0, stream>>>(qbf, membf, memT, att);

  // out = attended * Wo^T + bo + x   (f32 out)
  gemm_bf16<1, false><<<dim3(E_DIM / 128, N_ROWS / 128), 256, 0, stream>>>(
      att, wobf, bo, x, out, N_ROWS, E_DIM, E_DIM);
}
